// IPA3DRoPE_74191265071601
// MI455X (gfx1250) — matrix operand order repaired and checked by reading, not compiled, not run on hardware
//
#include <hip/hip_runtime.h>

#define NN     768
#define CS     384
#define CZ     128
#define NH     12
#define CC     16
#define INF_F  100000.0f
#define EPS_F  1e-08f
#define BASE_F 10000.0f

// 1/sqrt(3*C) and sqrt(1/3)
#define SCALE_QK 0.14433756729740643f
#define SQRT13   0.5773502691896258f

typedef __attribute__((ext_vector_type(16))) __bf16 v16bf;
typedef __attribute__((ext_vector_type(8)))  __bf16 v8bf;
typedef __attribute__((ext_vector_type(8)))  float  v8f;
typedef unsigned int u32x4 __attribute__((ext_vector_type(4)));
typedef int          i32x8 __attribute__((ext_vector_type(8)));
typedef int          i32x4 __attribute__((ext_vector_type(4)));

#if defined(__HIP_DEVICE_COMPILE__) && __has_builtin(__builtin_amdgcn_tensor_load_to_lds)
#define HAVE_TDM 1
#else
#define HAVE_TDM 0
#endif

__device__ __forceinline__ v8f wmma_bf16(v16bf a, v16bf b, v8f c) {
  // D = A(16x32 bf16) * B(32x16 bf16) + C(16x16 f32)
  return __builtin_amdgcn_wmma_f32_16x16x32_bf16(false, a, false, b, (short)0, c,
                                                 false, false);
}

__device__ __forceinline__ v16bf cat8(v8bf lo, v8bf hi) {
  return __builtin_shufflevector(lo, hi, 0, 1, 2, 3, 4, 5, 6, 7,
                                         8, 9, 10, 11, 12, 13, 14, 15);
}

#if HAVE_TDM
// TDM: async 2D tile load (128 x 16 rows, fp32) global -> LDS, per ISA D# layout.
__device__ __forceinline__ void tdm_load_tile(const float* gptr, unsigned ldsOff) {
  unsigned galo = (unsigned)__builtin_amdgcn_readfirstlane((int)(unsigned)((size_t)gptr));
  unsigned gahi = (unsigned)__builtin_amdgcn_readfirstlane((int)(unsigned)((size_t)gptr >> 32));
  unsigned lofs = (unsigned)__builtin_amdgcn_readfirstlane((int)ldsOff);
  u32x4 g0;
  g0[0] = 1u;                                   // count=1, no gather, user mode
  g0[1] = lofs;                                 // lds_addr (bytes)
  g0[2] = galo;                                 // global_addr[31:0]
  g0[3] = (gahi & 0x1ffffffu) | (2u << 30);     // global_addr[56:32] | type=2
  i32x8 g1;
  g1[0] = (int)(2u << 16);                      // data_size=4B, mask=0
  g1[1] = (int)((CZ & 0xffff) << 16);           // tensor_dim0 lo16 (=128)
  g1[2] = (int)(((NN * NN) & 0xffffu) << 16);   // dim0 hi16=0 | tensor_dim1 lo16
  g1[3] = (int)(((NN * NN) >> 16) | (CZ << 16));// tensor_dim1 hi16 | tile_dim0=128
  g1[4] = 16;                                   // tile_dim1=16, tile_dim2=0
  g1[5] = CZ;                                   // tensor_dim0_stride = 128
  g1[6] = 0;                                    // stride hi | dim1_stride lo
  g1[7] = 0;
  i32x4 g2 = {0, 0, 0, 0}, g3 = {0, 0, 0, 0};
  i32x8 g4 = {0, 0, 0, 0, 0, 0, 0, 0};
  __builtin_amdgcn_tensor_load_to_lds(g0, g1, g2, g3, g4, 0);
}
#endif

// ---------------------------------------------------------------------------
// K1: fp32 -> bf16 conversions (weights stored TRANSPOSED: Wt[col][k], so the
// WMMA B-operand becomes two contiguous 16B loads) + RoPE cos/sin tables.
// ---------------------------------------------------------------------------
__global__ void prep_kernel(const float* __restrict__ s, const float* __restrict__ rot,
                            const float* __restrict__ Wq, const float* __restrict__ Wkv,
                            const float* __restrict__ Wb, const float* __restrict__ Wout,
                            __bf16* s_bf, __bf16* Wq_t, __bf16* Wkv_t,
                            __bf16* Wb_t, __bf16* Wout_t,
                            float* cosT, float* sinT) {
  int tid = blockIdx.x * blockDim.x + threadIdx.x;
  int nth = gridDim.x * blockDim.x;
  for (int i = tid; i < NN * CS; i += nth) s_bf[i] = (__bf16)s[i];
  for (int i = tid; i < NH * CC * CS; i += nth) {           // Wq_t[col][k]
    int col = i / CS, k = i % CS;
    Wq_t[i] = (__bf16)Wq[k * (NH * CC) + col];
  }
  for (int i = tid; i < 2 * NH * CC * CS; i += nth) {       // Wkv_t[col][k]
    int col = i / CS, k = i % CS;
    Wkv_t[i] = (__bf16)Wkv[k * (2 * NH * CC) + col];
  }
  for (int i = tid; i < 16 * CZ; i += nth) {                // Wb_t[h][z], pad h->16
    int h = i >> 7, zc = i & 127;
    Wb_t[i] = (__bf16)((h < NH) ? Wb[zc * NH + h] : 0.0f);
  }
  for (int i = tid; i < CS * NH * CC; i += nth) {           // Wout_t[col][k]
    int col = i / (NH * CC), k = i % (NH * CC);
    Wout_t[i] = (__bf16)Wout[k * CS + col];
  }
  for (int i = tid; i < NN * 8; i += nth) {                 // C/2 = 8 rotary freqs
    int n = i >> 3, t = i & 7;
    const float* r = rot + n * 9;
    float tr  = r[0] + r[4] + r[8];
    float ca  = fminf(fmaxf((tr - 1.0f) * 0.5f, -1.0f + EPS_F), 1.0f - EPS_F);
    float ang = acosf(ca);
    float freq = __powf(BASE_F, -((float)(2 * t)) / 16.0f);
    float ph = ang * freq;
    cosT[i] = cosf(ph);
    sinT[i] = sinf(ph);
  }
}

// ---------------------------------------------------------------------------
// K2: q/k/v projection GEMM (K=384) + bias + fused RoPE.
// ---------------------------------------------------------------------------
__global__ void qkv_kernel(const __bf16* __restrict__ s_bf,
                           const __bf16* __restrict__ Wq_t,
                           const __bf16* __restrict__ Wkv_t,
                           const float* __restrict__ bq, const float* __restrict__ bkv,
                           const float* __restrict__ cosT, const float* __restrict__ sinT,
                           __bf16* q_bf, __bf16* k_bf, __bf16* v_t) {
  int wave = (blockIdx.x * blockDim.x + threadIdx.x) >> 5;
  int lane = threadIdx.x & 31;
  int it = wave / 36, ct = wave % 36;
  int i0 = it * 16;
  int m = lane & 15, kh = lane >> 4;

  const __bf16* Wt;
  int col;
  if (ct < 12) { Wt = Wq_t;  col = ct * 16 + m; }
  else         { Wt = Wkv_t; col = (ct - 12) * 16 + m; }
  const __bf16* wrow = Wt + (size_t)col * CS;

  v8f acc = {};
  for (int kk = 0; kk < 12; ++kk) {   // K = 384 = 12 * 32
    const __bf16* arow = s_bf + (size_t)(i0 + m) * CS + kk * 32;
    v16bf A = cat8(*(const v8bf*)(arow + (kh << 3)),
                   *(const v8bf*)(arow + 16 + (kh << 3)));
    const __bf16* brow = wrow + kk * 32 + (kh << 3);
    v16bf B = cat8(*(const v8bf*)brow, *(const v8bf*)(brow + 16));
    acc = wmma_bf16(A, B, acc);
  }

  float bias_c = (ct < 12) ? bq[col] : bkv[col];
#pragma unroll
  for (int r = 0; r < 8; ++r) acc[r] += bias_c;

  int idx = ct - 12;
  bool doRope = (ct < 12) || ((idx & 1) == 0);   // rope q and k, not v
  if (doRope) {
    int c2 = m >> 1;
#pragma unroll
    for (int r = 0; r < 8; ++r) {
      int row = i0 + (kh << 3) + r;
      float x = acc[r];
      float p = __shfl_xor(x, 1, 32);            // even/odd channel partner
      float cs = cosT[row * 8 + c2], sn = sinT[row * 8 + c2];
      acc[r] = (m & 1) ? (p * sn + x * cs) : (x * cs - p * sn);
    }
  }

#pragma unroll
  for (int r = 0; r < 8; ++r) {
    int row = i0 + (kh << 3) + r;
    __bf16 bv = (__bf16)acc[r];
    if (ct < 12) {
      q_bf[(size_t)row * (NH * CC) + ct * 16 + m] = bv;
    } else {
      int h = idx >> 1;
      if ((idx & 1) == 0) k_bf[(size_t)row * (NH * CC) + h * 16 + m] = bv;
      else                v_t[(size_t)(h * 16 + m) * NN + row] = bv;  // transposed
    }
  }
}

// ---------------------------------------------------------------------------
// K3: pair-bias GEMM  pb[h][i][j] = z[i,j,:] @ Wb  (M=N*N, K=128, N=16)
// HBM-bound: 302 MB of z. Each wave streams 8 consecutive 16-row tiles,
// double-buffered through LDS via the Tensor Data Mover when available.
// ---------------------------------------------------------------------------
__global__ void pb_kernel(const float* __restrict__ z,
                          const __bf16* __restrict__ Wb_t,
                          __bf16* __restrict__ pb_bf) {
  int w = threadIdx.x >> 5;
  int lane = threadIdx.x & 31;
  int m = lane & 15, kh = lane >> 4;
  int wave = blockIdx.x * (blockDim.x >> 5) + w;
  const int TPW = 8;                       // tiles (of 16 rows) per wave
  size_t tile0 = (size_t)wave * TPW;

#if HAVE_TDM
  __shared__ float zbuf[4][2][16 * CZ];    // 4 waves/block * 2 * 8KB = 64KB
  unsigned off0 = (unsigned)(size_t)&zbuf[w][0][0];
  unsigned off1 = (unsigned)(size_t)&zbuf[w][1][0];
  tdm_load_tile(z + tile0 * (16 * CZ), off0);
#endif

  for (int t = 0; t < TPW; ++t) {
#if HAVE_TDM
    if (t + 1 < TPW) {
      tdm_load_tile(z + (tile0 + t + 1) * (16 * CZ), (t & 1) ? off0 : off1);
      __builtin_amdgcn_s_wait_tensorcnt(1);   // tile t landed in LDS
    } else {
      __builtin_amdgcn_s_wait_tensorcnt(0);
    }
    const float* zr = &zbuf[w][t & 1][m * CZ];
#else
    const float* zr = z + (tile0 + t) * (16 * CZ) + (size_t)m * CZ;
#endif

    v8f acc = {};
#pragma unroll
    for (int kk = 0; kk < 4; ++kk) {       // K = 128 = 4 * 32
      const float* p1 = zr + kk * 32 + (kh << 3);
      const float* p2 = p1 + 16;
      v16bf A;
#pragma unroll
      for (int e = 0; e < 8; ++e) { A[e] = (__bf16)p1[e]; A[8 + e] = (__bf16)p2[e]; }
      const __bf16* wb = Wb_t + m * CZ + kk * 32 + (kh << 3);
      v16bf B = cat8(*(const v8bf*)wb, *(const v8bf*)(wb + 16));
      acc = wmma_bf16(A, B, acc);
    }

    if (m < NH) {                          // column = head (only 12 valid)
      size_t base = (tile0 + t) * 16;
#pragma unroll
      for (int r = 0; r < 8; ++r) {
        size_t ij = base + (kh << 3) + r;
        pb_bf[(size_t)m * ((size_t)NN * NN) + ij] = (__bf16)acc[r];
      }
    }
  }
}

// ---------------------------------------------------------------------------
// K4: fused attention, flash-style online softmax.
// One wave per (head, 16-row i-tile). 2 WMMAs per 16-wide j-tile.
// ---------------------------------------------------------------------------
__global__ void attn_kernel(const __bf16* __restrict__ q_bf,
                            const __bf16* __restrict__ k_bf,
                            const __bf16* __restrict__ v_t,
                            const __bf16* __restrict__ pb_bf,
                            const float* __restrict__ bb,
                            const float* __restrict__ mask,
                            __bf16* o_bf) {
  __shared__ __bf16 plds[8][16 * 24];      // 48B row stride: aligned b128 reload
  int w = threadIdx.x >> 5;
  int lane = threadIdx.x & 31;
  int wave = blockIdx.x * (blockDim.x >> 5) + w;
  int h = wave / 48, it = wave % 48;
  int i0 = it * 16;
  int m = lane & 15, kh = lane >> 4;
  v8bf z8 = {};

  // A-operand: q rows (K = C = 16, zero-padded to 32)
  v16bf Aq = cat8(*(const v8bf*)(q_bf + (size_t)(i0 + m) * (NH * CC) + h * 16 + (kh << 3)),
                  z8);

  float maskI[8];
#pragma unroll
  for (int r = 0; r < 8; ++r) maskI[r] = mask[i0 + (kh << 3) + r];
  float bbh = SQRT13 * bb[h];

  float mrun[8], lrun[8];
#pragma unroll
  for (int r = 0; r < 8; ++r) { mrun[r] = -1e30f; lrun[r] = 0.0f; }
  v8f acc = {};
  v8f zf = {};

  for (int j0 = 0; j0 < NN; j0 += 16) {
    // B-operand: k rows (K = c = 8*kh + s for s < 8; s >= 8 zero)
    const v8bf* kp = (const v8bf*)(k_bf + (size_t)(j0 + m) * (NH * CC) + h * 16);
    v16bf Bk = cat8(kp[kh], z8);
    v8f S = wmma_bf16(Aq, Bk, zf);

    float maskJ = mask[j0 + m];
#pragma unroll
    for (int r = 0; r < 8; ++r) {
      size_t irow = (size_t)(i0 + (kh << 3) + r);
      float pbv = (float)pb_bf[(size_t)h * ((size_t)NN * NN) + irow * NN + j0 + m];
      float s = S[r] * SCALE_QK + SQRT13 * pbv + bbh
              + INF_F * (maskI[r] * maskJ - 1.0f);
      float rmax = s;
      rmax = fmaxf(rmax, __shfl_xor(rmax, 1, 32));
      rmax = fmaxf(rmax, __shfl_xor(rmax, 2, 32));
      rmax = fmaxf(rmax, __shfl_xor(rmax, 4, 32));
      rmax = fmaxf(rmax, __shfl_xor(rmax, 8, 32));
      float mnew = fmaxf(mrun[r], rmax);
      float p = __expf(s - mnew);
      float rsum = p;
      rsum += __shfl_xor(rsum, 1, 32);
      rsum += __shfl_xor(rsum, 2, 32);
      rsum += __shfl_xor(rsum, 4, 32);
      rsum += __shfl_xor(rsum, 8, 32);
      float corr = __expf(mrun[r] - mnew);
      lrun[r] = lrun[r] * corr + rsum;
      mrun[r] = mnew;
      acc[r] *= corr;
      plds[w][(r + (kh << 3)) * 24 + m] = (__bf16)p;   // D-layout -> LDS
    }
    asm volatile("s_wait_dscnt 0" ::: "memory");       // P-tile visible to reload

    // A-operand: P in row-per-lane layout (K = j, zero-padded to 32)
    v16bf Ap = cat8(*(const v8bf*)&plds[w][m * 24 + (kh << 3)], z8);

    // B-operand: V^T rows (K = j = 8*kh + s for s < 8; s >= 8 zero, N = c)
    const v8bf* vp = (const v8bf*)(v_t + (size_t)(h * 16 + m) * NN + j0);
    v16bf Bv = cat8(vp[kh], z8);
    acc = wmma_bf16(Ap, Bv, acc);
  }

#pragma unroll
  for (int r = 0; r < 8; ++r) {
    float o = acc[r] / lrun[r];
    o_bf[(size_t)(i0 + (kh << 3) + r) * (NH * CC) + h * 16 + m] = (__bf16)o;
  }
}

// ---------------------------------------------------------------------------
// K5: output projection GEMM (K=192) + bias, fp32 result.
// ---------------------------------------------------------------------------
__global__ void out_kernel(const __bf16* __restrict__ o_bf,
                           const __bf16* __restrict__ Wout_t,
                           const float* __restrict__ bout,
                           float* __restrict__ out) {
  int wave = (blockIdx.x * blockDim.x + threadIdx.x) >> 5;
  int lane = threadIdx.x & 31;
  int it = wave / 24, ct = wave % 24;
  int i0 = it * 16;
  int m = lane & 15, kh = lane >> 4;
  int col = ct * 16 + m;
  const __bf16* wrow = Wout_t + (size_t)col * (NH * CC);

  v8f acc = {};
  for (int kk = 0; kk < 6; ++kk) {     // K = 192 = 6 * 32
    const __bf16* arow = o_bf + (size_t)(i0 + m) * (NH * CC) + kk * 32;
    v16bf A = cat8(*(const v8bf*)(arow + (kh << 3)),
                   *(const v8bf*)(arow + 16 + (kh << 3)));
    const __bf16* brow = wrow + kk * 32 + (kh << 3);
    v16bf B = cat8(*(const v8bf*)brow, *(const v8bf*)(brow + 16));
    acc = wmma_bf16(A, B, acc);
  }

  float bc = bout[col];
#pragma unroll
  for (int r = 0; r < 8; ++r)
    out[(size_t)(i0 + (kh << 3) + r) * CS + col] = acc[r] + bc;
}

// ---------------------------------------------------------------------------
extern "C" void kernel_launch(void* const* d_in, const int* in_sizes, int n_in,
                              void* d_out, int out_size, void* d_ws, size_t ws_size,
                              hipStream_t stream) {
  const float* s    = (const float*)d_in[0];
  const float* z    = (const float*)d_in[1];
  const float* rot  = (const float*)d_in[2];
  const float* mask = (const float*)d_in[3];
  const float* Wq   = (const float*)d_in[4];
  const float* bq   = (const float*)d_in[5];
  const float* Wkv  = (const float*)d_in[6];
  const float* bkv  = (const float*)d_in[7];
  const float* Wb   = (const float*)d_in[8];
  const float* bb   = (const float*)d_in[9];
  const float* Wout = (const float*)d_in[10];
  const float* bout = (const float*)d_in[11];
  float* out = (float*)d_out;

  char* w = (char*)d_ws;
  auto alloc = [&](size_t bytes) -> char* {
    char* p = w;
    w += (bytes + 255) & ~(size_t)255;
    return p;
  };
  __bf16* s_bf    = (__bf16*)alloc((size_t)NN * CS * 2);
  __bf16* Wq_t    = (__bf16*)alloc((size_t)NH * CC * CS * 2);
  __bf16* Wkv_t   = (__bf16*)alloc((size_t)2 * NH * CC * CS * 2);
  __bf16* Wb_t    = (__bf16*)alloc((size_t)16 * CZ * 2);
  __bf16* Wout_t  = (__bf16*)alloc((size_t)CS * NH * CC * 2);
  float*  cosT    = (float*)alloc((size_t)NN * 8 * 4);
  float*  sinT    = (float*)alloc((size_t)NN * 8 * 4);
  __bf16* q_bf    = (__bf16*)alloc((size_t)NN * NH * CC * 2);
  __bf16* k_bf    = (__bf16*)alloc((size_t)NN * NH * CC * 2);
  __bf16* v_t     = (__bf16*)alloc((size_t)NH * CC * NN * 2);
  __bf16* pb_bf   = (__bf16*)alloc((size_t)NH * NN * NN * 2);
  __bf16* o_bf    = (__bf16*)alloc((size_t)NN * NH * CC * 2);

  prep_kernel<<<256, 256, 0, stream>>>(s, rot, Wq, Wkv, Wb, Wout,
                                       s_bf, Wq_t, Wkv_t, Wb_t, Wout_t,
                                       cosT, sinT);
  // 48 i-tiles * 36 col-tiles = 1728 waves = 216 blocks of 8 waves
  qkv_kernel<<<216, 256, 0, stream>>>(s_bf, Wq_t, Wkv_t, bq, bkv, cosT, sinT,
                                      q_bf, k_bf, v_t);
  // 36864 tiles / 8 per wave = 4608 waves = 1152 blocks of 4 waves
  pb_kernel<<<1152, 128, 0, stream>>>(z, Wb_t, pb_bf);
  // 12 heads * 48 i-tiles = 576 waves = 72 blocks
  attn_kernel<<<72, 256, 0, stream>>>(q_bf, k_bf, v_t, pb_bf, bb, mask, o_bf);
  // 48 i-tiles * 24 col-tiles = 1152 waves = 144 blocks
  out_kernel<<<144, 256, 0, stream>>>(o_bf, Wout_t, bout, out);
}
